// RIM_2010044695291
// MI455X (gfx1250) — matrix operand order repaired and checked by reading, not compiled, not run on hardware
//
#include <hip/hip_runtime.h>
#include <hip/hip_bf16.h>
#include <stdint.h>

// ---------------------------------------------------------------------------
// MI455X (gfx1250) implementation.
// Eight 4096x1024x1024 GEMMs -> v_wmma_f32_16x16x32_f16 (wave32 WMMA),
// double-buffered LDS pipeline, f16 staging, f32 epilogue tail.
// ---------------------------------------------------------------------------

typedef __attribute__((ext_vector_type(16))) _Float16 v16h;
typedef __attribute__((ext_vector_type(8)))  float    v8f;

#define N_ROWS 4096
#define DDIM   1024
#define EDIM   1024
#define LDIM   128
#define ODIM   64
#define EPSK   1e-6f

// d_out layout: [0..63]=x_out, [64]=kl_g, [65]=kl_b,
//               [66..66+4095]=alpha_bars, [4162..5185]=H_g
#define OUT_KLG 64
#define OUT_KLB 65
#define OUT_AB  66
#define OUT_HG  (66 + N_ROWS)

// ------------------------------ device helpers ------------------------------

__device__ __forceinline__ float softplusf(float v) {
  return (v > 20.f) ? v : log1pf(expf(v));
}

__device__ __forceinline__ uint32_t hashu(uint32_t x) {
  x ^= x >> 16; x *= 0x7feb352du;
  x ^= x >> 15; x *= 0x846ca68bu;
  x ^= x >> 16; return x;
}

// stateless standard normal (Box-Muller on counter hash)
__device__ __forceinline__ float gaussf(uint32_t n, uint32_t salt) {
  uint32_t h1 = hashu(n * 0x9E3779B9u + salt);
  uint32_t h2 = hashu(h1 + 0x85EBCA6Bu);
  float u1 = ((h1 >> 8) + 0.5f) * (1.f / 16777216.f);
  float u2 = ((h2 >> 8) + 0.5f) * (1.f / 16777216.f);
  return sqrtf(-2.f * logf(u1)) * cosf(6.28318530718f * u2);
}

union FragH { v16h v; uint4 q[2]; };

// ------------------------------ WMMA GEMM ----------------------------------
// C[Ntot x Eout] = act(A[Ntot x K] @ B[K x Eout] + bias[Eout])
// block tile 128x64, 8 wave32s; wave w owns rows 16w..16w+15 and 4 n-subtiles.
// 2-stage LDS double buffer: one barrier per K-step; global loads for tile
// k+1 overlap the 4 back-to-back v_wmma of tile k.
__global__ __launch_bounds__(256)
void gemm_wmma(const _Float16* __restrict__ A, const _Float16* __restrict__ B,
               const float* __restrict__ bias, _Float16* __restrict__ C,
               int K, int Eout, int act) {
  __shared__ _Float16 a_sh[2][128][48];   // 32 K-halves + pad, 16B-aligned rows
  __shared__ _Float16 b_sh[2][64][48];    // transposed: [n][k]

  const int t    = threadIdx.x;
  const int lane = t & 31, w = t >> 5;
  const int lm   = lane & 15, lh = lane >> 4;
  const int row0 = blockIdx.y * 128;
  const int col0 = blockIdx.x * 64;

  // staging assignments
  const int ar = t >> 1,        ac = (t & 1) * 16;   // A: row, 16-half chunk
  const int bk = t >> 3,        bn = (t & 7) * 8;    // B: k-row, 8-n chunk

  v8f acc[4] = {};

  // ---- prologue: stage tile 0 into buffer 0 ----
  {
    const _Float16* Ag = A + (size_t)(row0 + ar) * K + ac;
    uint4 v0 = *(const uint4*)Ag;
    uint4 v1 = *(const uint4*)(Ag + 8);
    *(uint4*)&a_sh[0][ar][ac]     = v0;
    *(uint4*)&a_sh[0][ar][ac + 8] = v1;
    const _Float16* Bg = B + (size_t)bk * Eout + (col0 + bn);
    uint4 bv = *(const uint4*)Bg;
    _Float16 tmp[8];
    *(uint4*)tmp = bv;
#pragma unroll
    for (int i = 0; i < 8; ++i) b_sh[0][bn + i][bk] = tmp[i];
  }
  __syncthreads();

  int buf = 0;
  for (int k0 = 0; k0 < K; k0 += 32) {
    // ---- kick off global loads + LDS stores for the NEXT tile ----
    if (k0 + 32 < K) {
      const _Float16* Ag = A + (size_t)(row0 + ar) * K + (k0 + 32 + ac);
      uint4 v0 = *(const uint4*)Ag;
      uint4 v1 = *(const uint4*)(Ag + 8);
      *(uint4*)&a_sh[buf ^ 1][ar][ac]     = v0;
      *(uint4*)&a_sh[buf ^ 1][ar][ac + 8] = v1;
      const _Float16* Bg = B + (size_t)(k0 + 32 + bk) * Eout + (col0 + bn);
      uint4 bv = *(const uint4*)Bg;
      _Float16 tmp[8];
      *(uint4*)tmp = bv;
#pragma unroll
      for (int i = 0; i < 8; ++i) b_sh[buf ^ 1][bn + i][bk] = tmp[i];
    }
    if (k0 + 64 < K) {  // warm the K-slab after next (global_prefetch_b8)
      __builtin_prefetch(A + (size_t)(row0 + ar) * K + k0 + 64, 0, 1);
      __builtin_prefetch(B + (size_t)(k0 + 64 + bk) * Eout + col0, 0, 1);
    }

    // ---- preload ALL fragments, then burst 4 WMMAs back-to-back ----
    // A fragment per ISA 16-bit A layout: lane-half selects k-octets {0,8}+{0,16}
    FragH af;
    af.q[0] = *(const uint4*)&a_sh[buf][16 * w + lm][8 * lh];
    af.q[1] = *(const uint4*)&a_sh[buf][16 * w + lm][16 + 8 * lh];
    // B fragments: lane holds column n=lm, k-octets {0,8}+{0,16} matching A
    FragH bf[4];
#pragma unroll
    for (int j = 0; j < 4; ++j) {
      bf[j].q[0] = *(const uint4*)&b_sh[buf][16 * j + lm][8 * lh];
      bf[j].q[1] = *(const uint4*)&b_sh[buf][16 * j + lm][16 + 8 * lh];
    }
#pragma unroll
    for (int j = 0; j < 4; ++j) {
      acc[j] = __builtin_amdgcn_wmma_f32_16x16x32_f16(
          false, af.v, false, bf[j].v, (short)0, acc[j], false, false);
    }

    __syncthreads();   // next-tile stores done AND this-tile reads consumed
    buf ^= 1;
  }

  // epilogue: C layout vgpr r, lane l -> (m = r + 8*(l/16), n = l%16)
#pragma unroll
  for (int j = 0; j < 4; ++j) {
#pragma unroll
    for (int r = 0; r < 8; ++r) {
      int m = 16 * w + r + 8 * lh;
      int n = col0 + 16 * j + lm;
      float v = acc[j][r] + bias[n];
      if (act == 1) v = fmaxf(v, 0.0f);
      C[(size_t)(row0 + m) * Eout + n] = (_Float16)v;
    }
  }
}

// ------------------------------ small kernels -------------------------------

__global__ void f32_to_f16(const float* __restrict__ s, _Float16* __restrict__ d, int n) {
  int i = blockIdx.x * 256 + threadIdx.x;
  if (i < n) d[i] = (_Float16)s[i];
}

__global__ void init_out_kernel(float* d_out) {
  if (threadIdx.x < 2) d_out[OUT_KLG + threadIdx.x] = 0.f;
}

// xv[e] = ge_b1[e] + sum_d x[d] * ge_W1[D+d][e]   (rank-1 row-constant bias)
__global__ void xv_kernel(const float* __restrict__ x, const float* __restrict__ geW1,
                          const float* __restrict__ geb1, float* __restrict__ xv) {
  int e = blockIdx.x * 256 + threadIdx.x;
  float acc = geb1[e];
  for (int d = 0; d < DDIM; ++d) acc += x[d] * geW1[(size_t)(DDIM + d) * EDIM + e];
  xv[e] = acc;
}

// out[row] = act(dot(M[row], w) + bptr[bidx]); one wave32 per row
// act: 0 none, 1 softplus, 2 0.4*sigmoid
__global__ __launch_bounds__(256)
void rowdot(const _Float16* __restrict__ M, const float* __restrict__ w,
            const float* __restrict__ bptr, int bidx,
            float* __restrict__ out, int E, int act) {
  int lane = threadIdx.x & 31;
  int wave = threadIdx.x >> 5;
  int row  = blockIdx.x * 8 + wave;
  const _Float16* mr = M + (size_t)row * E;
  float acc = 0.f;
  for (int k = lane; k < E; k += 32) acc += (float)mr[k] * w[k];
#pragma unroll
  for (int off = 16; off > 0; off >>= 1) acc += __shfl_xor(acc, off, 32);
  if (lane == 0) {
    float v = acc + bptr[bidx];
    if (act == 1)      v = softplusf(v);
    else if (act == 2) v = 0.4f / (1.f + expf(-v));
    out[row] = v;
  }
}

// per-row sampling + kl_b accumulation
__global__ __launch_bounds__(256)
void scalar_stage(const float* __restrict__ prior_mij, const float* __restrict__ pma,
                  const float* __restrict__ psa, const float* __restrict__ post_mean,
                  const float* __restrict__ post_std,
                  float* __restrict__ ab_raw, float* __restrict__ at_arr,
                  float* __restrict__ sat_arr, float* __restrict__ a1_arr,
                  float* __restrict__ a2sq_arr, float* __restrict__ t2_arr,
                  float* d_out) {
  __shared__ float red[256];
  int n = blockIdx.x * 256 + threadIdx.x;
  float nij   = softplusf(pma[n]) + 0.01f;
  float ps    = psa[n];
  float nij_  = 2.f * nij * ps * ps;
  float pmij  = 0.5f * (1.f + nij_ - sqrtf(nij_ * nij_ + 1.f));
  float stda  = sqrtf(fmaxf(pmij * (1.f - pmij), 0.f));
  float e1    = gaussf((uint32_t)n, 0x12345687u);
  float e2    = gaussf((uint32_t)n, 0x89ABCDEFu);
  float at    = softplusf(e1 * stda + pmij);
  float sat   = sqrtf(at);
  float pstd  = post_std[n], pmean = post_mean[n];
  float sij   = 0.01f * e2 * sat * pstd + at * pmean;
  ab_raw[n]   = sij * at;
  at_arr[n]   = at;
  sat_arr[n]  = sat;
  a1_arr[n]   = at * pmean;
  float a2    = sat * pstd;
  a2sq_arr[n] = a2 * a2;
  t2_arr[n]   = 2.f * logf(a2 + EPSK);
  float pm    = prior_mij[n];
  float klb   = fabsf(pm - pmij + pmij * (logf(pmij + EPSK) - logf(pm + EPSK)));
  red[threadIdx.x] = klb;
  __syncthreads();
  for (int s = 128; s > 0; s >>= 1) {
    if (threadIdx.x < s) red[threadIdx.x] += red[threadIdx.x + s];
    __syncthreads();
  }
  if (threadIdx.x == 0) atomicAdd(d_out + OUT_KLB, red[0]);
}

// single-block min/max + normalize -> d_out alpha_bars
__global__ __launch_bounds__(1024)
void minmax_norm(const float* __restrict__ ab_raw, float* __restrict__ d_out) {
  __shared__ float smin[1024], smax[1024];
  int t = threadIdx.x;
  float mn = 1e30f, mx = -1e30f;
  for (int n = t; n < N_ROWS; n += 1024) {
    float v = ab_raw[n];
    mn = fminf(mn, v); mx = fmaxf(mx, v);
  }
  smin[t] = mn; smax[t] = mx;
  __syncthreads();
  for (int s = 512; s > 0; s >>= 1) {
    if (t < s) { smin[t] = fminf(smin[t], smin[t + s]); smax[t] = fmaxf(smax[t], smax[t + s]); }
    __syncthreads();
  }
  float lo = smin[0], inv = 1.f / (smax[0] - smin[0]);
  for (int n = t; n < N_ROWS; n += 1024) d_out[OUT_AB + n] = (ab_raw[n] - lo) * inv;
}

// kl_g: block per i, loop j; j-arrays live in L2 (192MB)
__global__ __launch_bounds__(256)
void klg_kernel(const float* __restrict__ prior_mean, const float* __restrict__ prior_std,
                const float* __restrict__ at_arr, const float* __restrict__ sat_arr,
                const float* __restrict__ a1_arr, const float* __restrict__ a2sq_arr,
                const float* __restrict__ t2_arr, float* d_out) {
  __shared__ float red[256];
  int i = blockIdx.x;
  float pm = prior_mean[i], ps = prior_std[i];
  float s = 0.f;
  for (int j = threadIdx.x; j < N_ROWS; j += 256) {
    float at   = at_arr[j];
    float a4e  = ps * sat_arr[j] + EPSK;
    float diff = a1_arr[j] - pm * at;
    float r = 2.f * logf(a4e) - t2_arr[j]
            + (a2sq_arr[j] + diff * diff) / (a4e * a4e) - 1.f;
    s += fabsf(r);
  }
  red[threadIdx.x] = s;
  __syncthreads();
  for (int q = 128; q > 0; q >>= 1) {
    if (threadIdx.x < q) red[threadIdx.x] += red[threadIdx.x + q];
    __syncthreads();
  }
  if (threadIdx.x == 0) atomicAdd(d_out + OUT_KLG, 0.5f * red[0]);
}

// H_g raw: hg[d] = sum_n alpha_bars[n] * base[n][d]
__global__ void hg_kernel(const float* __restrict__ d_out, const float* __restrict__ base,
                          float* __restrict__ hg_raw) {
  int d = blockIdx.x * 256 + threadIdx.x;
  const float* ab = d_out + OUT_AB;
  float acc = 0.f;
  for (int n = 0; n < N_ROWS; ++n) acc += ab[n] * base[(size_t)n * DDIM + d];
  hg_raw[d] = acc;
}

__global__ __launch_bounds__(1024)
void finalize_hg(const float* __restrict__ hg_raw, const float* __restrict__ x,
                 float* __restrict__ d_out, float* __restrict__ hgf) {
  __shared__ float red[1024];
  int t = threadIdx.x;
  float v = hg_raw[t];
  red[t] = fabsf(v);
  __syncthreads();
  for (int s = 512; s > 0; s >>= 1) {
    if (t < s) red[t] += red[t + s];
    __syncthreads();
  }
  float h = 0.5f * v / red[0] + 0.5f * x[t];
  d_out[OUT_HG + t] = h;
  hgf[t] = h;
}

__global__ __launch_bounds__(128)
void out_mlp(const float* __restrict__ hgf, const float* __restrict__ W1,
             const float* __restrict__ b1, const float* __restrict__ W2,
             const float* __restrict__ b2, float* __restrict__ d_out) {
  __shared__ float y1[LDIM];
  int t = threadIdx.x;
  float acc = b1[t];
  for (int d = 0; d < DDIM; ++d) acc += hgf[d] * W1[(size_t)d * LDIM + t];
  y1[t] = acc;
  __syncthreads();
  if (t < ODIM) {
    float a = b2[t];
    for (int e = 0; e < LDIM; ++e) a += y1[e] * W2[(size_t)e * ODIM + t];
    d_out[t] = a;
  }
}

// --------------------------------- launch ----------------------------------

extern "C" void kernel_launch(void* const* d_in, const int* in_sizes, int n_in,
                              void* d_out_v, int out_size, void* d_ws, size_t ws_size,
                              hipStream_t stream) {
  (void)in_sizes; (void)n_in; (void)out_size; (void)ws_size;

  const float* x        = (const float*)d_in[0];
  const float* base     = (const float*)d_in[1];
  const float* ge_W1    = (const float*)d_in[2];
  const float* ge_b1    = (const float*)d_in[3];
  const float* ge_W2    = (const float*)d_in[4];
  const float* ge_b2    = (const float*)d_in[5];
  const float* enc_W    = (const float*)d_in[6];
  const float* enc_b    = (const float*)d_in[7];
  const float* encg_W   = (const float*)d_in[8];
  const float* encg_b   = (const float*)d_in[9];
  const float* encb_W   = (const float*)d_in[10];
  const float* encb_b   = (const float*)d_in[11];
  const float* mean_W   = (const float*)d_in[12];
  const float* mean_b   = (const float*)d_in[13];
  const float* std_W    = (const float*)d_in[14];
  const float* std_b    = (const float*)d_in[15];
  const float* pm_W     = (const float*)d_in[16];
  const float* pm_b     = (const float*)d_in[17];
  const float* pma_W    = (const float*)d_in[18];
  const float* pma_b    = (const float*)d_in[19];
  const float* psa_W    = (const float*)d_in[20];
  const float* psa_b    = (const float*)d_in[21];
  const float* out_W1   = (const float*)d_in[22];
  const float* out_b1   = (const float*)d_in[23];
  const float* out_W2   = (const float*)d_in[24];
  const float* out_b2   = (const float*)d_in[25];
  float* d_out = (float*)d_out_v;

  // ---- workspace carve (256B aligned) ----
  char* p = (char*)d_ws;
  auto alloc = [&](size_t bytes) -> void* {
    void* r = (void*)p;
    p += (bytes + 255) & ~(size_t)255;
    return r;
  };
  const size_t NE2 = (size_t)N_ROWS * EDIM * sizeof(_Float16);
  const size_t EE2 = (size_t)EDIM * EDIM * sizeof(_Float16);

  _Float16* base_h  = (_Float16*)alloc((size_t)N_ROWS * DDIM * 2);
  _Float16* W1a_h   = (_Float16*)alloc(EE2);
  _Float16* W2_h    = (_Float16*)alloc(EE2);
  _Float16* encW_h  = (_Float16*)alloc(2 * EE2);
  _Float16* encgW_h = (_Float16*)alloc(2 * EE2);
  _Float16* encbW_h = (_Float16*)alloc(2 * EE2);
  _Float16* h1      = (_Float16*)alloc(NE2);
  _Float16* edge    = (_Float16*)alloc(NE2);
  _Float16* hbuf    = (_Float16*)alloc(NE2);
  _Float16* hgbuf   = (_Float16*)alloc(NE2);
  _Float16* hbbuf   = (_Float16*)alloc(NE2);
  float* xv         = (float*)alloc(EDIM * 4);
  float* prior_mean = (float*)alloc(N_ROWS * 4);
  float* prior_std  = (float*)alloc(N_ROWS * 4);
  float* post_mean  = (float*)alloc(N_ROWS * 4);
  float* post_std   = (float*)alloc(N_ROWS * 4);
  float* prior_mij  = (float*)alloc(N_ROWS * 4);
  float* pma        = (float*)alloc(N_ROWS * 4);
  float* psa        = (float*)alloc(N_ROWS * 4);
  float* ab_raw     = (float*)alloc(N_ROWS * 4);
  float* at_arr     = (float*)alloc(N_ROWS * 4);
  float* sat_arr    = (float*)alloc(N_ROWS * 4);
  float* a1_arr     = (float*)alloc(N_ROWS * 4);
  float* a2sq_arr   = (float*)alloc(N_ROWS * 4);
  float* t2_arr     = (float*)alloc(N_ROWS * 4);
  float* hg_raw     = (float*)alloc(DDIM * 4);
  float* hgf        = (float*)alloc(DDIM * 4);

  // ---- 0: zero atomic accumulators ----
  init_out_kernel<<<1, 64, 0, stream>>>(d_out);

  // ---- 1: f32 -> f16 staging ----
  auto conv = [&](const float* s, _Float16* d, int n) {
    f32_to_f16<<<(n + 255) / 256, 256, 0, stream>>>(s, d, n);
  };
  conv(base, base_h, N_ROWS * DDIM);
  conv(ge_W1, W1a_h, DDIM * EDIM);            // rows 0..D-1 of ge_W1
  conv(ge_W2, W2_h, EDIM * EDIM);
  conv(enc_W, encW_h, 2 * EDIM * EDIM);
  conv(encg_W, encgW_h, 2 * EDIM * EDIM);
  conv(encb_W, encbW_h, 2 * EDIM * EDIM);

  // ---- 2: xv = x @ ge_W1[D:] + ge_b1 ----
  xv_kernel<<<EDIM / 256, 256, 0, stream>>>(x, ge_W1, ge_b1, xv);

  const dim3 ggrid(EDIM / 64, N_ROWS / 128);
  // ---- 3: h1 = relu(base @ W1a + xv) ----
  gemm_wmma<<<ggrid, 256, 0, stream>>>(base_h, W1a_h, xv, h1, DDIM, EDIM, 1);
  // ---- 4: edge = h1 @ ge_W2 + ge_b2 ----
  gemm_wmma<<<ggrid, 256, 0, stream>>>(h1, W2_h, ge_b2, edge, EDIM, EDIM, 0);

  // ---- 5: encode branches ----
  for (int i = 0; i < 2; ++i) {
    const size_t EE = (size_t)EDIM * EDIM;
    gemm_wmma<<<ggrid, 256, 0, stream>>>(edge, encW_h + i * EE, enc_b + i * EDIM,
                                         hbuf, EDIM, EDIM, 1);
    gemm_wmma<<<ggrid, 256, 0, stream>>>(hbuf, encgW_h + i * EE, encg_b + i * EDIM,
                                         hgbuf, EDIM, EDIM, 1);
    gemm_wmma<<<ggrid, 256, 0, stream>>>(hbuf, encbW_h + i * EE, encb_b + i * EDIM,
                                         hbbuf, EDIM, EDIM, 1);
    float* mean_o = i ? post_mean : prior_mean;
    float* std_o  = i ? post_std  : prior_std;
    rowdot<<<N_ROWS / 8, 256, 0, stream>>>(hgbuf, mean_W + i * EDIM, mean_b, i,
                                           mean_o, EDIM, 0);
    rowdot<<<N_ROWS / 8, 256, 0, stream>>>(hgbuf, std_W + i * EDIM, std_b, i,
                                           std_o, EDIM, 1);
    if (i == 0) {
      rowdot<<<N_ROWS / 8, 256, 0, stream>>>(hbbuf, pm_W, pm_b, 0, prior_mij, EDIM, 2);
    } else {
      rowdot<<<N_ROWS / 8, 256, 0, stream>>>(hbbuf, pma_W, pma_b, 0, pma, EDIM, 0);
      rowdot<<<N_ROWS / 8, 256, 0, stream>>>(hbbuf, psa_W, psa_b, 0, psa, EDIM, 1);
    }
  }

  // ---- 6: per-row posterior + sampling + kl_b ----
  scalar_stage<<<N_ROWS / 256, 256, 0, stream>>>(prior_mij, pma, psa, post_mean,
                                                 post_std, ab_raw, at_arr, sat_arr,
                                                 a1_arr, a2sq_arr, t2_arr, d_out);
  // ---- 7: alpha_bars normalize -> d_out ----
  minmax_norm<<<1, 1024, 0, stream>>>(ab_raw, d_out);
  // ---- 8: kl_g (N^2) ----
  klg_kernel<<<N_ROWS, 256, 0, stream>>>(prior_mean, prior_std, at_arr, sat_arr,
                                         a1_arr, a2sq_arr, t2_arr, d_out);
  // ---- 9: H_g aggregation, normalize + residual, output MLP ----
  hg_kernel<<<DDIM / 256, 256, 0, stream>>>(d_out, base, hg_raw);
  finalize_hg<<<1, 1024, 0, stream>>>(hg_raw, x, d_out, hgf);
  out_mlp<<<1, 128, 0, stream>>>(hgf, out_W1, out_b1, out_W2, out_b2, d_out);
}
